// E3NN_PhaseNet_simple_54692113547902
// MI455X (gfx1250) — matrix operand order repaired and checked by reading, not compiled, not run on hardware
//
#include <hip/hip_runtime.h>
#include <cmath>

// ---------------------------------------------------------------------------
// Problem constants (from reference)
// ---------------------------------------------------------------------------
static constexpr int NN = 20000;         // nodes
static constexpr int NE = 320000;        // edges
static constexpr int NUM_BASIS = 16;
static constexpr float MAX_RADIUS = 3.15f;

// CG table layout (12 triples, row-major [i][j][k] each)
static constexpr int CG_FLOATS = 11148;

__host__ __device__ constexpr int cg_off(int l1, int l2, int l3) {
  return (l1 == 0 && l2 == 4) ? 0 : (l1 == 0 && l2 == 6) ? 81
       : (l1 == 4 && l2 == 4 && l3 == 4) ? 250  : (l1 == 4 && l2 == 4 && l3 == 6) ? 979
       : (l1 == 4 && l2 == 6 && l3 == 4) ? 2032 : (l1 == 4 && l2 == 6 && l3 == 6) ? 3085
       : (l1 == 6 && l2 == 4 && l3 == 4) ? 4606 : (l1 == 6 && l2 == 4 && l3 == 6) ? 5659
       : (l1 == 6 && l2 == 6 && l3 == 4) ? 7180 : (l1 == 6 && l2 == 6 && l3 == 6) ? 8701
       : (l1 == 4 && l2 == 4 && l3 == 0) ? 10898 : 10979;
}

__host__ __device__ constexpr double cfact(int n) {
  double r = 1.0; for (int i = 2; i <= n; ++i) r *= i; return r;
}
__host__ __device__ constexpr double cdfact(int n) {
  double r = 1.0; while (n > 1) { r *= n; n -= 2; } return r;
}

// ---------------------------------------------------------------------------
// CDNA5 WMMA (wave32): f16 inputs, f32 accumulate
// ---------------------------------------------------------------------------
typedef __attribute__((ext_vector_type(16))) _Float16 v16h;
typedef __attribute__((ext_vector_type(8)))  _Float16 v8h;
typedef __attribute__((ext_vector_type(8)))  float    v8f;

__device__ __forceinline__ v8f wmma_f16(v16h a, v16h b, v8f c) {
  return __builtin_amdgcn_wmma_f32_16x16x32_f16(false, a, false, b, (short)0, c,
                                                false, false);
}
__device__ __forceinline__ v16h cat8(v8h lo, v8h hi) {
  return __builtin_shufflevector(lo, hi, 0, 1, 2, 3, 4, 5, 6, 7,
                                 8, 9, 10, 11, 12, 13, 14, 15);
}
// fast native reciprocal (v_rcp_f32) and silu
__device__ __forceinline__ float fast_rcp(float x) { return __builtin_amdgcn_rcpf(x); }
__device__ __forceinline__ float fast_silu(float x) {
  return x * fast_rcp(1.f + __expf(-x));
}

// ---------------------------------------------------------------------------
// Device CG computation (runs once per launch; 12 threads, double precision)
// ---------------------------------------------------------------------------
struct cpx { double re, im; };
__device__ __forceinline__ cpx cmul(cpx a, cpx b) {
  return {a.re * b.re - a.im * b.im, a.re * b.im + a.im * b.re};
}
__device__ __forceinline__ cpx cadd(cpx a, cpx b) { return {a.re + b.re, a.im + b.im}; }

__device__ double dev_fact(int n) { double r = 1.0; for (int i = 2; i <= n; ++i) r *= i; return r; }

__device__ double dev_cg(int j1, int m1, int j2, int m2, int j3, int m3) {
  if (m1 + m2 != m3) return 0.0;
  int dj = j1 - j2; if (dj < 0) dj = -dj;
  if (j3 < dj || j3 > j1 + j2) return 0.0;
  double pre = sqrt((2.0 * j3 + 1.0) * dev_fact(j1 + j2 - j3) * dev_fact(j1 - j2 + j3) *
                    dev_fact(-j1 + j2 + j3) / dev_fact(j1 + j2 + j3 + 1));
  pre *= sqrt(dev_fact(j1 + m1) * dev_fact(j1 - m1) * dev_fact(j2 + m2) *
              dev_fact(j2 - m2) * dev_fact(j3 + m3) * dev_fact(j3 - m3));
  int k0 = 0;
  if (j2 - j3 - m1 > k0) k0 = j2 - j3 - m1;
  if (j1 - j3 + m2 > k0) k0 = j1 - j3 + m2;
  int k1 = j1 + j2 - j3;
  if (j1 - m1 < k1) k1 = j1 - m1;
  if (j2 + m2 < k1) k1 = j2 + m2;
  double s = 0.0;
  for (int k = k0; k <= k1; ++k) {
    double term = 1.0 / (dev_fact(k) * dev_fact(j1 + j2 - j3 - k) * dev_fact(j1 - m1 - k) *
                         dev_fact(j2 + m2 - k) * dev_fact(j3 - j2 + m1 + k) *
                         dev_fact(j3 - j1 - m2 + k));
    s += (k & 1) ? -term : term;
  }
  return pre * s;
}

__device__ void dev_makeU(int l, cpx* U) {
  int d = 2 * l + 1;
  for (int i = 0; i < d * d; ++i) U[i] = {0.0, 0.0};
  U[l * d + l] = {1.0, 0.0};
  const double s2 = 0.7071067811865476;
  for (int m = 1; m <= l; ++m) {
    double sg = (m & 1) ? -1.0 : 1.0;
    U[(l + m) * d + (l + m)] = {sg * s2, 0.0};
    U[(l + m) * d + (l - m)] = {s2, 0.0};
    U[(l - m) * d + (l + m)] = {0.0, -sg * s2};
    U[(l - m) * d + (l - m)] = {0.0, s2};
  }
}

__global__ void cg_kernel(float* __restrict__ cgout, double* __restrict__ scratch) {
  const int t = threadIdx.x;
  if (t >= 12) return;
  const int L1[12] = {0, 0, 4, 4, 4, 4, 6, 6, 6, 6, 4, 6};
  const int L2[12] = {4, 6, 4, 4, 6, 6, 4, 4, 6, 6, 4, 6};
  const int L3[12] = {4, 6, 4, 6, 4, 6, 4, 6, 4, 6, 0, 0};
  const int OFF[12] = {0, 81, 250, 979, 2032, 3085, 4606, 5659, 7180, 8701, 10898, 10979};
  const int l1 = L1[t], l2 = L2[t], l3 = L3[t];
  const int d1 = 2 * l1 + 1, d2 = 2 * l2 + 1, d3 = 2 * l3 + 1;
  const int MAXE = 2197;
  cpx* W  = (cpx*)scratch + (size_t)t * 3 * MAXE;
  cpx* T1 = W + MAXE;
  cpx* T2 = W + 2 * MAXE;
  cpx U1[169], U2[169], U3[169];
  dev_makeU(l1, U1); dev_makeU(l2, U2); dev_makeU(l3, U3);
  const int n = d1 * d2 * d3;
  for (int i = 0; i < n; ++i) W[i] = {0.0, 0.0};
  for (int m1 = -l1; m1 <= l1; ++m1)
    for (int m2 = -l2; m2 <= l2; ++m2) {
      int m3 = -(m1 + m2);
      if (m3 < -l3 || m3 > l3) continue;
      int p = l1 - l2 - m3;
      double sg = (p & 1) ? -1.0 : 1.0;
      W[((m1 + l1) * d2 + (m2 + l2)) * d3 + (m3 + l3)] =
          {sg / sqrt(2.0 * l3 + 1.0) * dev_cg(l1, m1, l2, m2, l3, -m3), 0.0};
    }
  for (int a = 0; a < d1; ++a)
    for (int n2 = 0; n2 < d2; ++n2)
      for (int p = 0; p < d3; ++p) {
        cpx acc = {0.0, 0.0};
        for (int m = 0; m < d1; ++m)
          acc = cadd(acc, cmul(U1[a * d1 + m], W[(m * d2 + n2) * d3 + p]));
        T1[(a * d2 + n2) * d3 + p] = acc;
      }
  for (int a = 0; a < d1; ++a)
    for (int b = 0; b < d2; ++b)
      for (int p = 0; p < d3; ++p) {
        cpx acc = {0.0, 0.0};
        for (int n2 = 0; n2 < d2; ++n2)
          acc = cadd(acc, cmul(U2[b * d2 + n2], T1[(a * d2 + n2) * d3 + p]));
        T2[(a * d2 + b) * d3 + p] = acc;
      }
  for (int a = 0; a < d1; ++a)
    for (int b = 0; b < d2; ++b)
      for (int c = 0; c < d3; ++c) {
        cpx acc = {0.0, 0.0};
        for (int p = 0; p < d3; ++p)
          acc = cadd(acc, cmul(U3[c * d3 + p], T2[(a * d2 + b) * d3 + p]));
        W[(a * d2 + b) * d3 + c] = acc;
      }
  double nr = 0.0, ni = 0.0;
  for (int i = 0; i < n; ++i) { nr += W[i].re * W[i].re; ni += W[i].im * W[i].im; }
  bool ur = (sqrt(nr) >= sqrt(ni));
  double norm = sqrt(ur ? nr : ni);
  double sc = sqrt(2.0 * l3 + 1.0) / norm;
  for (int i = 0; i < n; ++i)
    cgout[OFF[t] + i] = (float)((ur ? W[i].re : W[i].im) * sc);
}

// ---------------------------------------------------------------------------
// Weight prep: W (K x N, f32) -> WT (N x Kpad, f16, zero-padded K..Kpad)
// ---------------------------------------------------------------------------
__global__ void convert_wT(const float* __restrict__ W, _Float16* __restrict__ WT,
                           int K, int Kpad, int Ncols) {
  int i = blockIdx.x * blockDim.x + threadIdx.x;
  if (i >= Ncols * Kpad) return;
  int col = i / Kpad, k = i - col * Kpad;
  WT[i] = (k < K) ? (_Float16)W[k * Ncols + col] : (_Float16)0.0f;
}

// ---------------------------------------------------------------------------
// Spherical harmonics (l=4 -> sh[0..8], l=6 -> sh[9..21])
// ---------------------------------------------------------------------------
__device__ __forceinline__ void compute_sh(float x, float y, float z, float (&sh)[22]) {
  float A[7], B[7];
  A[0] = 1.f; B[0] = 0.f;
#pragma unroll
  for (int m = 1; m < 7; ++m) {
    A[m] = A[m - 1] * x - B[m - 1] * y;
    B[m] = A[m - 1] * y + B[m - 1] * x;
  }
  float Q[7][7];
#pragma unroll
  for (int m = 0; m < 7; ++m) {
    Q[m][m] = (float)cdfact(2 * m - 1);
    if (m + 1 < 7) Q[m + 1][m] = z * (float)((2 * m + 1) * cdfact(2 * m - 1));
#pragma unroll
    for (int ll = m + 2; ll < 7; ++ll)
      Q[ll][m] = ((2.f * ll - 1.f) * z * Q[ll - 1][m] - (float)(ll + m - 1) * Q[ll - 2][m]) *
                 (1.f / (float)(ll - m));
  }
#pragma unroll
  for (int m = -4; m <= 4; ++m) {
    int am = m < 0 ? -m : m;
    float c = sqrtf((m != 0 ? 2.f : 1.f) * 9.f * (float)(cfact(4 - am) / cfact(4 + am)));
    sh[m + 4] = c * Q[4][am] * ((m < 0) ? B[am] : A[am]);
  }
#pragma unroll
  for (int m = -6; m <= 6; ++m) {
    int am = m < 0 ? -m : m;
    float c = sqrtf((m != 0 ? 2.f : 1.f) * 13.f * (float)(cfact(6 - am) / cfact(6 + am)));
    sh[9 + m + 6] = c * Q[6][am] * ((m < 0) ? B[am] : A[am]);
  }
}

// ---------------------------------------------------------------------------
// One tensor-product path: matmul2 (WMMA) -> wS, then per-lane CG contraction
// accumulated into the caller's register accumulator `acc` (one irrep group).
// ---------------------------------------------------------------------------
template <int IN_DIM, int MU1, int L1, int INOFF, int L2, int MU3, int L3, int WOFF>
__device__ __forceinline__ void do_path(int lane, const float* __restrict__ hin, int src,
                                        const _Float16* mlpS, float* wS,
                                        const _Float16* __restrict__ W2T,
                                        const float* __restrict__ cg,
                                        const float (&sh)[22],
                                        float (&acc)[MU3 * (2 * L3 + 1)]) {
  constexpr int WP = MU1 * MU3;     // path weight block width
  constexpr int NT = WP / 16;       // N tiles of matmul2
  constexpr int D1 = 2 * L1 + 1, D2 = 2 * L2 + 1, D3 = 2 * L3 + 1;
  constexpr int SHOFF = (L2 == 4) ? 0 : 9;
  const int lh = lane & 15, hh = lane >> 4;

  // --- matmul2: silu-features (32 x 64) @ W2T[WOFF..WOFF+WP][64] ---
#pragma unroll
  for (int nt = 0; nt < NT; ++nt) {
    const v8h* pb0 = (const v8h*)(W2T + (WOFF + nt * 16 + lh) * 64 + hh * 8);
    const v8h* pb1 = (const v8h*)(W2T + (WOFF + nt * 16 + lh) * 64 + 32 + hh * 8);
    v16h b0 = cat8(pb0[0], pb0[2]);
    v16h b1 = cat8(pb1[0], pb1[2]);
#pragma unroll
    for (int mt = 0; mt < 2; ++mt) {
      const v8h* pa0 = (const v8h*)(mlpS + (mt * 16 + lh) * 64 + hh * 8);
      const v8h* pa1 = (const v8h*)(mlpS + (mt * 16 + lh) * 64 + 32 + hh * 8);
      v16h a0 = cat8(pa0[0], pa0[2]);
      v16h a1 = cat8(pa1[0], pa1[2]);
      v8f c = {0.f, 0.f, 0.f, 0.f, 0.f, 0.f, 0.f, 0.f};
      c = wmma_f16(a0, b0, c);
      c = wmma_f16(a1, b1, c);
#pragma unroll
      for (int r = 0; r < 8; ++r)
        wS[(mt * 16 + r + hh * 8) * WP + nt * 16 + lh] = c[r] * 0.125f;  // /sqrt(64)
    }
  }
  __syncthreads();

  // --- per-lane CG tensor product; lane owns edge `lane` of the tile ---
  const float* C = cg + cg_off(L1, L2, L3);
  for (int i = 0; i < D1; ++i) {
    float xs[MU1];
#pragma unroll
    for (int u = 0; u < MU1; ++u) xs[u] = hin[src * IN_DIM + INOFF + u * D1 + i];
    float xw[MU3];
#pragma unroll
    for (int v = 0; v < MU3; ++v) {
      float t = 0.f;
#pragma unroll
      for (int u = 0; u < MU1; ++u) t += xs[u] * wS[lane * WP + u * MU3 + v];
      xw[v] = t;
    }
    float sc[D3];
#pragma unroll
    for (int k = 0; k < D3; ++k) {
      float t = 0.f;
#pragma unroll
      for (int j = 0; j < D2; ++j) t += sh[SHOFF + j] * C[(i * D2 + j) * D3 + k];
      sc[k] = t;
    }
#pragma unroll
    for (int v = 0; v < MU3; ++v)
#pragma unroll
      for (int k = 0; k < D3; ++k) acc[v * D3 + k] += xw[v] * sc[k];
  }
  __syncthreads();
}

template <int OUT_DIM, int MU3, int D3, int OUTOFF>
__device__ __forceinline__ void writeback(const float* acc, float* __restrict__ hout,
                                          int dst, float scale) {
#pragma unroll
  for (int v = 0; v < MU3; ++v)
#pragma unroll
    for (int k = 0; k < D3; ++k)
      unsafeAtomicAdd(&hout[dst * OUT_DIM + OUTOFF + v * D3 + k], acc[v * D3 + k] * scale);
}

// ---------------------------------------------------------------------------
// Fused conv layer: one wave32 workgroup handles a tile of 32 edges.
// ---------------------------------------------------------------------------
template <int LAYER>
__global__ __launch_bounds__(32) void conv_kernel(
    const float* __restrict__ hin, float* __restrict__ hout,
    const int* __restrict__ edge_index, const float* __restrict__ edge_vec,
    const _Float16* __restrict__ W1T, const _Float16* __restrict__ W2T,
    const float* __restrict__ cg) {
  constexpr int IN_DIM  = (LAYER == 0) ? 4 : 176;
  constexpr int OUT_DIM = (LAYER == 2) ? 4 : 176;

  __shared__ alignas(16) _Float16 hbS[32 * 32];   // zero-padded K: 16..31 = 0
  __shared__ alignas(16) _Float16 mlpS[32 * 64];
  __shared__ alignas(16) float    wS[32 * 64];

  const int lane = threadIdx.x;
  const int lh = lane & 15, hh = lane >> 4;
  const int e = blockIdx.x * 32 + lane;
  const int src = edge_index[e];
  const int dst = edge_index[NE + e];

  // geometry: distance, radial basis, spherical harmonics
  const float vx = edge_vec[3 * e + 0], vy = edge_vec[3 * e + 1], vz = edge_vec[3 * e + 2];
  const float d = sqrtf(vx * vx + vy * vy + vz * vz);
  const float inv = fast_rcp(fmaxf(d, 1e-9f));
  float sh[22];
  compute_sh(vx * inv, vy * inv, vz * inv, sh);

  const float inv_estep = (float)(NUM_BASIS + 1) / MAX_RADIUS;
  const float estep = MAX_RADIUS / (float)(NUM_BASIS + 1);
  const float envc = 1.14136f * 7.3890560989f * 4.0f;  // 1.14136*e^2*sqrt(NUM_BASIS)
#pragma unroll
  for (int b = 0; b < NUM_BASIS; ++b) {
    float t = (d - (float)(b + 1) * estep) * inv_estep;
    float tp = t + 1.f, tm = 1.f - t;
    float f1 = tp > 0.f ? __expf(-fast_rcp(tp)) : 0.f;
    float f2 = tm > 0.f ? __expf(-fast_rcp(tm)) : 0.f;
    hbS[lane * 32 + b] = (_Float16)(envc * f1 * f2);
    hbS[lane * 32 + 16 + b] = (_Float16)0.0f;      // zero pad K=16..31
  }
  __syncthreads();

  // --- matmul1: hb(32x16, K padded to 32) @ W1T, scale, silu ---
#pragma unroll
  for (int mt = 0; mt < 2; ++mt) {
#pragma unroll
    for (int nt = 0; nt < 4; ++nt) {
      const v8h* pa = (const v8h*)(hbS + (mt * 16 + lh) * 32 + hh * 8);
      const v8h* pb = (const v8h*)(W1T + (nt * 16 + lh) * 32 + hh * 8);
      v16h a = cat8(pa[0], pa[2]);
      v16h b = cat8(pb[0], pb[2]);
      v8f c = {0.f, 0.f, 0.f, 0.f, 0.f, 0.f, 0.f, 0.f};
      c = wmma_f16(a, b, c);
#pragma unroll
      for (int r = 0; r < 8; ++r) {
        int m = r + hh * 8;
        float yv = fast_silu(c[r] * 0.25f);    // /sqrt(16) then silu
        mlpS[(mt * 16 + m) * 64 + nt * 16 + lh] = (_Float16)yv;
      }
    }
  }
  __syncthreads();

  // --- tensor-product paths grouped by destination irrep (register acc) ---
  constexpr float fanscale = (LAYER == 0) ? 0.5f
                           : (LAYER == 1) ? 0.17677669529663687f  // 1/sqrt(32)
                                          : 0.25f;                // 1/sqrt(16)
  const float scale = fanscale * 0.28867513459481287f;            // 1/sqrt(12)

  if constexpr (LAYER == 0) {
    {
      float acc[8 * 9] = {};
      do_path<IN_DIM, 4, 0, 0, 4, 8, 4, 0>(lane, hin, src, mlpS, wS, W2T, cg, sh, acc);
      writeback<OUT_DIM, 8, 9, 0>(acc, hout, dst, scale);
    }
    {
      float acc[8 * 13] = {};
      do_path<IN_DIM, 4, 0, 0, 6, 8, 6, 32>(lane, hin, src, mlpS, wS, W2T, cg, sh, acc);
      writeback<OUT_DIM, 8, 13, 72>(acc, hout, dst, scale);
    }
  } else if constexpr (LAYER == 1) {
    {
      float acc[8 * 9] = {};
      do_path<IN_DIM, 8, 4, 0,  4, 8, 4, 0>(lane, hin, src, mlpS, wS, W2T, cg, sh, acc);
      do_path<IN_DIM, 8, 4, 0,  6, 8, 4, 128>(lane, hin, src, mlpS, wS, W2T, cg, sh, acc);
      do_path<IN_DIM, 8, 6, 72, 4, 8, 4, 256>(lane, hin, src, mlpS, wS, W2T, cg, sh, acc);
      do_path<IN_DIM, 8, 6, 72, 6, 8, 4, 384>(lane, hin, src, mlpS, wS, W2T, cg, sh, acc);
      writeback<OUT_DIM, 8, 9, 0>(acc, hout, dst, scale);
    }
    {
      float acc[8 * 13] = {};
      do_path<IN_DIM, 8, 4, 0,  4, 8, 6, 64>(lane, hin, src, mlpS, wS, W2T, cg, sh, acc);
      do_path<IN_DIM, 8, 4, 0,  6, 8, 6, 192>(lane, hin, src, mlpS, wS, W2T, cg, sh, acc);
      do_path<IN_DIM, 8, 6, 72, 4, 8, 6, 320>(lane, hin, src, mlpS, wS, W2T, cg, sh, acc);
      do_path<IN_DIM, 8, 6, 72, 6, 8, 6, 448>(lane, hin, src, mlpS, wS, W2T, cg, sh, acc);
      writeback<OUT_DIM, 8, 13, 72>(acc, hout, dst, scale);
    }
  } else {
    float acc[4] = {};
    do_path<IN_DIM, 8, 4, 0,  4, 4, 0, 0>(lane, hin, src, mlpS, wS, W2T, cg, sh, acc);
    do_path<IN_DIM, 8, 6, 72, 6, 4, 0, 32>(lane, hin, src, mlpS, wS, W2T, cg, sh, acc);
    writeback<OUT_DIM, 4, 1, 0>(acc, hout, dst, scale);
  }
}

// ---------------------------------------------------------------------------
// Utility kernels
// ---------------------------------------------------------------------------
__global__ void zero_kernel(float* __restrict__ p, int n) {
  int i = blockIdx.x * blockDim.x + threadIdx.x;
  if (i < n) p[i] = 0.f;
}

__global__ void init_h0_kernel(const int* __restrict__ x, const float* __restrict__ emb,
                               float* __restrict__ h0) {
  int i = blockIdx.x * blockDim.x + threadIdx.x;
  if (i < NN * 4) h0[i] = emb[x[i >> 2] * 4 + (i & 3)];
}

__global__ void head_kernel(const float* __restrict__ h3, const float* __restrict__ w1,
                            const float* __restrict__ w2, float* __restrict__ out) {
  int n = blockIdx.x * blockDim.x + threadIdx.x;
  if (n >= NN) return;
  float h[4];
#pragma unroll
  for (int c = 0; c < 4; ++c) h[c] = h3[n * 4 + c];
  float o[4] = {0.f, 0.f, 0.f, 0.f};
  for (int j = 0; j < 64; ++j) {
    float t = 0.f;
#pragma unroll
    for (int c = 0; c < 4; ++c) t += h[c] * w1[c * 64 + j];
    t = t * 0.5f;                              // /sqrt(4)
    t = t * __builtin_amdgcn_rcpf(1.f + __expf(-t));  // silu
#pragma unroll
    for (int k = 0; k < 4; ++k) o[k] += t * w2[j * 4 + k];
  }
#pragma unroll
  for (int k = 0; k < 4; ++k) out[n * 4 + k] = o[k] * 0.125f;  // /sqrt(64)
#pragma unroll
  for (int c = 0; c < 4; ++c) out[NN * 4 + n * 4 + c] = h[c];  // second tuple element
}

// ---------------------------------------------------------------------------
// Launch
// ---------------------------------------------------------------------------
extern "C" void kernel_launch(void* const* d_in, const int* in_sizes, int n_in,
                              void* d_out, int out_size, void* d_ws, size_t ws_size,
                              hipStream_t stream) {
  (void)in_sizes; (void)n_in; (void)out_size; (void)ws_size;
  const int*   x     = (const int*)d_in[0];
  const int*   ei    = (const int*)d_in[1];
  const float* evec  = (const float*)d_in[2];
  const float* emb   = (const float*)d_in[3];
  const float* c0w1  = (const float*)d_in[4];
  const float* c0w2  = (const float*)d_in[5];
  const float* c1w1  = (const float*)d_in[6];
  const float* c1w2  = (const float*)d_in[7];
  const float* c2w1  = (const float*)d_in[8];
  const float* c2w2  = (const float*)d_in[9];
  const float* hw1   = (const float*)d_in[10];
  const float* hw2   = (const float*)d_in[11];

  float* ws = (float*)d_ws;
  float* cg = ws;                       // CG_FLOATS
  float* h0 = cg + CG_FLOATS;           // NN*4
  float* h1 = h0 + NN * 4;              // NN*176
  float* h2 = h1 + NN * 176;            // NN*176
  float* h3 = h2 + NN * 176;            // NN*4

  // f16 transposed/padded weights (region is 16-byte aligned by construction)
  _Float16* w1t0 = (_Float16*)(h3 + NN * 4);      // 64 x 32
  _Float16* w1t1 = w1t0 + 64 * 32;
  _Float16* w1t2 = w1t1 + 64 * 32;
  _Float16* w2t0 = w1t2 + 64 * 32;                // 64  x 64
  _Float16* w2t1 = w2t0 + 64 * 64;                // 512 x 64
  _Float16* w2t2 = w2t1 + 512 * 64;               // 64  x 64

  // CG coefficients (double-precision scratch reuses h1 region; h1 is zeroed
  // afterwards on the same stream, so ordering is safe).
  cg_kernel<<<1, 32, 0, stream>>>(cg, (double*)h1);

  // Weight prep (f32 -> transposed, zero-padded f16)
  convert_wT<<<(64 * 32 + 255) / 256, 256, 0, stream>>>(c0w1, w1t0, 16, 32, 64);
  convert_wT<<<(64 * 32 + 255) / 256, 256, 0, stream>>>(c1w1, w1t1, 16, 32, 64);
  convert_wT<<<(64 * 32 + 255) / 256, 256, 0, stream>>>(c2w1, w1t2, 16, 32, 64);
  convert_wT<<<(64 * 64 + 255) / 256, 256, 0, stream>>>(c0w2, w2t0, 64, 64, 64);
  convert_wT<<<(512 * 64 + 255) / 256, 256, 0, stream>>>(c1w2, w2t1, 64, 64, 512);
  convert_wT<<<(64 * 64 + 255) / 256, 256, 0, stream>>>(c2w2, w2t2, 64, 64, 64);

  init_h0_kernel<<<(NN * 4 + 255) / 256, 256, 0, stream>>>(x, emb, h0);

  zero_kernel<<<(NN * 176 + 255) / 256, 256, 0, stream>>>(h1, NN * 176);
  conv_kernel<0><<<NE / 32, 32, 0, stream>>>(h0, h1, ei, evec, w1t0, w2t0, cg);

  zero_kernel<<<(NN * 176 + 255) / 256, 256, 0, stream>>>(h2, NN * 176);
  conv_kernel<1><<<NE / 32, 32, 0, stream>>>(h1, h2, ei, evec, w1t1, w2t1, cg);

  zero_kernel<<<(NN * 4 + 255) / 256, 256, 0, stream>>>(h3, NN * 4);
  conv_kernel<2><<<NE / 32, 32, 0, stream>>>(h2, h3, ei, evec, w1t2, w2t2, cg);

  head_kernel<<<(NN + 255) / 256, 256, 0, stream>>>(h3, hw1, hw2, (float*)d_out);
}
